// CausalSelfAttention_38517266711193
// MI455X (gfx1250) — matrix operand order repaired and checked by reading, not compiled, not run on hardware
//
#include <hip/hip_runtime.h>
#include <hip/hip_bf16.h>

// ---------------------------------------------------------------------------
// CausalSelfAttention (RoPE d=21, 9 heads, head_dim 64) for gfx1250 / MI455X.
// All matmuls via v_wmma_f32_16x16x32_bf16 (bf16 operands, f32 accum).
// ---------------------------------------------------------------------------

typedef __bf16 bf16_t;
typedef __bf16 v8bf  __attribute__((ext_vector_type(8)));
typedef __bf16 v16bf __attribute__((ext_vector_type(16)));
typedef float  v8f   __attribute__((ext_vector_type(8)));

#define B_SZ 4
#define T_SEQ 2048
#define C_DIM 576
#define H_CNT 9
#define KV_D 21
#define KV_FLAT 189
#define KV_PAD 192
#define BT (B_SZ * T_SEQ)

// ---- WMMA helpers ---------------------------------------------------------

__device__ __forceinline__ v8f wmma_bf16(v16bf a, v16bf b, v8f c) {
  // (neg_a, A, neg_b, B, c_mod, C, reuse_a, reuse_b)
  return __builtin_amdgcn_wmma_f32_16x16x32_bf16(false, a, false, b, (short)0, c,
                                                 false, false);
}

// A-matrix 16x32 bf16 fragment (row-major source).
// Lane l holds row m = l&15; g = l>>4 selects K chunks [g*8, g*8+8) and
// [16+g*8, 16+g*8+8) per the CDNA5 16-bit A layout.
__device__ __forceinline__ v16bf load_a16(const bf16_t* rowp, int g) {
  v8bf lo = *(const v8bf*)(rowp + g * 8);
  v8bf hi = *(const v8bf*)(rowp + 16 + g * 8);
  return __builtin_shufflevector(lo, hi, 0, 1, 2, 3, 4, 5, 6, 7, 8, 9, 10, 11,
                                 12, 13, 14, 15);
}

// B-matrix 32x16 bf16 fragment from N x K row-major ("Bt") storage.
// Lane l holds column n = l&15, K chunks [g*8, g*8+8) and
// [16+g*8, 16+g*8+8), matching the A layout.
__device__ __forceinline__ v16bf load_b16(const bf16_t* np_row, int g) {
  v8bf lo = *(const v8bf*)(np_row + g * 8);
  v8bf hi = *(const v8bf*)(np_row + 16 + g * 8);
  return __builtin_shufflevector(lo, hi, 0, 1, 2, 3, 4, 5, 6, 7, 8, 9, 10, 11,
                                 12, 13, 14, 15);
}

// ---- small utility kernels ------------------------------------------------

__global__ void cvt_f32_bf16_kernel(const float* __restrict__ src,
                                    bf16_t* __restrict__ dst, int n) {
  int i = blockIdx.x * blockDim.x + threadIdx.x;
  int stride = gridDim.x * blockDim.x;
  for (; i < n; i += stride) dst[i] = (bf16_t)src[i];
}

__global__ void fill_zero_bf16_kernel(bf16_t* __restrict__ dst, int n) {
  int i = blockIdx.x * blockDim.x + threadIdx.x;
  int stride = gridDim.x * blockDim.x;
  for (; i < n; i += stride) dst[i] = (bf16_t)0.0f;
}

// Wt[n][k] = W[k][n] as bf16, rows n >= N zero-filled (pad to Npad).
__global__ void transpose_w_kernel(const float* __restrict__ W,
                                   bf16_t* __restrict__ Wt, int K, int N,
                                   int Npad) {
  int idx = blockIdx.x * blockDim.x + threadIdx.x;
  if (idx >= Npad * K) return;
  int n = idx / K;
  int k = idx - n * K;
  Wt[(size_t)n * K + k] = (n < N) ? (bf16_t)W[(size_t)k * N + n] : (bf16_t)0.0f;
}

// ---- generic bf16 WMMA GEMM: C(MxN,f32) = A(MxK) * B(KxN), B given as
// Bt = B^T stored N x K row-major. One wave computes a 16x64 tile.
// Requires M%16==0, N%64==0, K%32==0; grid*8 waves == (M/16)*(N/64).
__global__ void gemm_bf16_wmma_kernel(const bf16_t* __restrict__ A,
                                      const bf16_t* __restrict__ Bt,
                                      float* __restrict__ C, int M, int N,
                                      int K) {
  int wid = blockIdx.x * (blockDim.x >> 5) + (threadIdx.x >> 5);
  int lane = threadIdx.x & 31;
  int g = lane >> 4;
  int ln = lane & 15;
  int ntiles_n = N >> 6;
  int mt = wid / ntiles_n;
  int nt = wid - mt * ntiles_n;
  if (mt * 16 >= M) return;

  const bf16_t* arow = A + (size_t)(mt * 16 + ln) * K;
  v8f acc[4] = {v8f{}, v8f{}, v8f{}, v8f{}};

  for (int kk = 0; kk < K; kk += 32) {
    v16bf a = load_a16(arow + kk, g);
#pragma unroll
    for (int j = 0; j < 4; ++j) {
      const bf16_t* brow = Bt + (size_t)(nt * 64 + j * 16 + ln) * K + kk;
      v16bf b = load_b16(brow, g);
      acc[j] = wmma_bf16(a, b, acc[j]);
    }
  }
  // C layout: element r of v8f -> row m = g*8 + r, col n = ln.
#pragma unroll
  for (int j = 0; j < 4; ++j) {
#pragma unroll
    for (int r = 0; r < 8; ++r) {
      C[(size_t)(mt * 16 + g * 8 + r) * N + nt * 64 + j * 16 + ln] = acc[j][r];
    }
  }
}

// ---- RoPE (odd dim 21) + pack into WMMA-friendly padded layouts -----------
// qrot/krot: [b][h][t][32] bf16 (d 21..31 zero).  vt: [b][h][32][T] bf16.
__global__ void rope_pack_kernel(const float* __restrict__ pq,
                                 const float* __restrict__ pk,
                                 const float* __restrict__ pv,
                                 bf16_t* __restrict__ qrot,
                                 bf16_t* __restrict__ krot,
                                 bf16_t* __restrict__ vt) {
  int idx = blockIdx.x * blockDim.x + threadIdx.x;
  if (idx >= B_SZ * H_CNT * T_SEQ) return;
  int t = idx & (T_SEQ - 1);
  int h = (idx >> 11) % H_CNT;
  int b = idx / (H_CNT * T_SEQ);

  const float* qp = pq + (size_t)(b * T_SEQ + t) * C_DIM + h * 64;
  const float* kp = pk + (size_t)(b * T_SEQ + t) * KV_PAD + h * KV_D;
  const float* vp = pv + (size_t)(b * T_SEQ + t) * KV_PAD + h * KV_D;

  float qv[KV_D], kv[KV_D];
#pragma unroll
  for (int j = 0; j < KV_D; ++j) { qv[j] = qp[j]; kv[j] = kp[j]; }

  // inv_freq[i] = 10000^(-(2i)/21), i = 0..10 ; emb = concat(freqs,freqs)[:21]
  float invf[11];
#pragma unroll
  for (int i = 0; i < 11; ++i)
    invf[i] = __expf(-(2.0f * (float)i / 21.0f) * 9.210340371976184f);

  bf16_t* qo = qrot + ((size_t)(b * H_CNT + h) * T_SEQ + t) * 32;
  bf16_t* ko = krot + ((size_t)(b * H_CNT + h) * T_SEQ + t) * 32;
#pragma unroll
  for (int j = 0; j < KV_D; ++j) {
    int fi = (j < 11) ? j : j - 11;
    float ang = (float)t * invf[fi];
    float c = cosf(ang);
    float s = sinf(ang);
    // rotate_half for length 21 (h=10): rot[j] = (j<11) ? -x[j+10] : x[j-11]
    float qr = (j < 11) ? -qv[j + 10] : qv[j - 11];
    float kr = (j < 11) ? -kv[j + 10] : kv[j - 11];
    qo[j] = (bf16_t)(qv[j] * c + qr * s);
    ko[j] = (bf16_t)(kv[j] * c + kr * s);
  }
#pragma unroll
  for (int j = KV_D; j < 32; ++j) { qo[j] = (bf16_t)0.0f; ko[j] = (bf16_t)0.0f; }

  bf16_t* vo = vt + (size_t)(b * H_CNT + h) * 32 * T_SEQ + t;
#pragma unroll
  for (int j = 0; j < 32; ++j)
    vo[(size_t)j * T_SEQ] = (j < KV_D) ? (bf16_t)vp[j] : (bf16_t)0.0f;
}

// ---- flash attention: one wave = one (b, h, 16-query tile) ----------------
__global__ void attn_kernel(const bf16_t* __restrict__ qrot,
                            const bf16_t* __restrict__ krot,
                            const bf16_t* __restrict__ vt,
                            bf16_t* __restrict__ ypad) {
  __shared__ __align__(32) bf16_t psmem[4][16 * 32];  // probs tile per wave
  int wslot = threadIdx.x >> 5;
  int lane = threadIdx.x & 31;
  int g = lane >> 4;
  int ln = lane & 15;
  int wid = blockIdx.x * 4 + wslot;
  int qt = wid & 127;               // T/16 = 128 query tiles
  int h = (wid >> 7) % H_CNT;
  int b = wid / (128 * H_CNT);

  const bf16_t* qbase = qrot + (size_t)(b * H_CNT + h) * T_SEQ * 32;
  const bf16_t* kbase = krot + (size_t)(b * H_CNT + h) * T_SEQ * 32;
  const bf16_t* vbase = vt + (size_t)(b * H_CNT + h) * 32 * T_SEQ;

  // Q fragment (16 queries x 32 padded d), loaded once.
  v16bf qa = load_a16(qbase + (size_t)(qt * 16 + ln) * 32, g);

  float Mx[8], Sm[8];
  v8f o0 = v8f{};  // output cols 0..15
  v8f o1 = v8f{};  // output cols 16..31
#pragma unroll
  for (int r = 0; r < 8; ++r) { Mx[r] = -1e30f; Sm[r] = 0.0f; }

  const int qlast = qt * 16 + 15;
  const float scale = 0.125f;  // 1/sqrt(64)

  for (int kb = 0; kb <= qlast; kb += 32) {
    // scores: K dim = 32 padded d ; two 16-key tiles
    v16bf kb0 = load_b16(kbase + (size_t)(kb + ln) * 32, g);
    v16bf kb1 = load_b16(kbase + (size_t)(kb + 16 + ln) * 32, g);
    v8f s0 = wmma_bf16(qa, kb0, v8f{});
    v8f s1 = wmma_bf16(qa, kb1, v8f{});

#pragma unroll
    for (int r = 0; r < 8; ++r) {
      int m = g * 8 + r;
      int qg = qt * 16 + m;
      float e0 = (kb + ln <= qg) ? s0[r] * scale : -1e9f;
      float e1 = (kb + 16 + ln <= qg) ? s1[r] * scale : -1e9f;
      float mx = fmaxf(e0, e1);
      mx = fmaxf(mx, __shfl_xor(mx, 1, 32));
      mx = fmaxf(mx, __shfl_xor(mx, 2, 32));
      mx = fmaxf(mx, __shfl_xor(mx, 4, 32));
      mx = fmaxf(mx, __shfl_xor(mx, 8, 32));
      float nM = fmaxf(Mx[r], mx);
      float alpha = __expf(Mx[r] - nM);
      float p0 = __expf(e0 - nM);
      float p1 = __expf(e1 - nM);
      float rs = p0 + p1;
      rs += __shfl_xor(rs, 1, 32);
      rs += __shfl_xor(rs, 2, 32);
      rs += __shfl_xor(rs, 4, 32);
      rs += __shfl_xor(rs, 8, 32);
      Sm[r] = Sm[r] * alpha + rs;
      Mx[r] = nM;
      o0[r] *= alpha;
      o1[r] *= alpha;
      // stash probs (C layout -> LDS 16x32 row-major tile)
      psmem[wslot][m * 32 + ln] = (bf16_t)p0;
      psmem[wslot][m * 32 + 16 + ln] = (bf16_t)p1;
    }
    // per-wave LDS ops are in-order; wait stores, then reload in A layout
    asm volatile("s_wait_dscnt 0" ::: "memory");
    v16bf pa = load_a16(&psmem[wslot][ln * 32], g);
    // PV: K dim = 32 keys ; vT rows are output dims
    v16bf vb0 = load_b16(vbase + (size_t)ln * T_SEQ + kb, g);
    v16bf vb1 = load_b16(vbase + (size_t)(16 + ln) * T_SEQ + kb, g);
    o0 = wmma_bf16(pa, vb0, o0);
    o1 = wmma_bf16(pa, vb1, o1);
  }

#pragma unroll
  for (int r = 0; r < 8; ++r) {
    int m = g * 8 + r;
    float inv = 1.0f / Sm[r];
    size_t row = (size_t)(b * T_SEQ + qt * 16 + m) * C_DIM + h * 64;
    ypad[row + ln] = (bf16_t)(o0[r] * inv);
    ypad[row + 16 + ln] = (bf16_t)(o1[r] * inv);
  }
}

// ---------------------------------------------------------------------------

extern "C" void kernel_launch(void* const* d_in, const int* in_sizes, int n_in,
                              void* d_out, int out_size, void* d_ws,
                              size_t ws_size, hipStream_t stream) {
  const float* x = (const float*)d_in[0];
  const float* Wq = (const float*)d_in[1];
  const float* Wk = (const float*)d_in[2];
  const float* Wv = (const float*)d_in[3];
  const float* Wo = (const float*)d_in[4];
  float* out = (float*)d_out;

  char* ws = (char*)d_ws;
  size_t off = 0;
  auto alloc = [&](size_t bytes) {
    size_t p = off;
    off += (bytes + 255) & ~(size_t)255;
    return p;
  };

  bf16_t* xb = (bf16_t*)(ws + alloc((size_t)BT * C_DIM * 2));
  bf16_t* WqT = (bf16_t*)(ws + alloc((size_t)C_DIM * C_DIM * 2));
  bf16_t* WkT = (bf16_t*)(ws + alloc((size_t)KV_PAD * C_DIM * 2));
  bf16_t* WvT = (bf16_t*)(ws + alloc((size_t)KV_PAD * C_DIM * 2));
  bf16_t* WoT = (bf16_t*)(ws + alloc((size_t)C_DIM * C_DIM * 2));
  float* projq = (float*)(ws + alloc((size_t)BT * C_DIM * 4));
  float* projk = (float*)(ws + alloc((size_t)BT * KV_PAD * 4));
  float* projv = (float*)(ws + alloc((size_t)BT * KV_PAD * 4));
  bf16_t* qrot = (bf16_t*)(ws + alloc((size_t)B_SZ * H_CNT * T_SEQ * 32 * 2));
  bf16_t* krot = (bf16_t*)(ws + alloc((size_t)B_SZ * H_CNT * T_SEQ * 32 * 2));
  bf16_t* vtb = (bf16_t*)(ws + alloc((size_t)B_SZ * H_CNT * 32 * T_SEQ * 2));
  bf16_t* ypad = (bf16_t*)(ws + alloc((size_t)BT * C_DIM * 2));
  (void)ws_size; (void)in_sizes; (void)n_in; (void)out_size;

  // 1) fp32 -> bf16 activation, transposed bf16 weights
  cvt_f32_bf16_kernel<<<1024, 256, 0, stream>>>(x, xb, BT * C_DIM);
  transpose_w_kernel<<<(C_DIM * C_DIM + 255) / 256, 256, 0, stream>>>(
      Wq, WqT, C_DIM, C_DIM, C_DIM);
  transpose_w_kernel<<<(KV_PAD * C_DIM + 255) / 256, 256, 0, stream>>>(
      Wk, WkT, C_DIM, KV_FLAT, KV_PAD);
  transpose_w_kernel<<<(KV_PAD * C_DIM + 255) / 256, 256, 0, stream>>>(
      Wv, WvT, C_DIM, KV_FLAT, KV_PAD);
  transpose_w_kernel<<<(C_DIM * C_DIM + 255) / 256, 256, 0, stream>>>(
      Wo, WoT, C_DIM, C_DIM, C_DIM);

  // 2) Q/K/V projections (WMMA GEMMs)
  // waves = (M/16)*(N/64); 8 waves per 256-thread block
  gemm_bf16_wmma_kernel<<<(BT / 16) * (C_DIM / 64) / 8, 256, 0, stream>>>(
      xb, WqT, projq, BT, C_DIM, C_DIM);
  gemm_bf16_wmma_kernel<<<(BT / 16) * (KV_PAD / 64) / 8, 256, 0, stream>>>(
      xb, WkT, projk, BT, KV_PAD, C_DIM);
  gemm_bf16_wmma_kernel<<<(BT / 16) * (KV_PAD / 64) / 8, 256, 0, stream>>>(
      xb, WvT, projv, BT, KV_PAD, C_DIM);

  // 3) RoPE + pack to padded WMMA layouts
  rope_pack_kernel<<<(B_SZ * H_CNT * T_SEQ + 255) / 256, 256, 0, stream>>>(
      projq, projk, projv, qrot, krot, vtb);

  // 4) zero the padded y buffer (head cols 32..63 stay zero)
  fill_zero_bf16_kernel<<<1024, 256, 0, stream>>>(ypad, BT * C_DIM);

  // 5) causal flash attention (WMMA scores + WMMA PV)
  attn_kernel<<<(B_SZ * H_CNT * (T_SEQ / 16)) / 4, 128, 0, stream>>>(
      qrot, krot, vtb, ypad);

  // 6) output projection: out = ypad @ Wo
  gemm_bf16_wmma_kernel<<<(BT / 16) * (C_DIM / 64) / 8, 256, 0, stream>>>(
      ypad, WoT, out, BT, C_DIM, C_DIM);
}
